// Trans4map_deformable_detr_plus_74440373174915
// MI455X (gfx1250) — compile-verified, matrix operand order checked by reading, not run on hardware
//
#include <hip/hip_runtime.h>
#include <hip/hip_bf16.h>
#include <climits>

// ---------------- problem constants ----------------
#define BEV_H 200
#define BEV_W 200
#define ED    128
#define HEADS 8
#define HD    16
#define LEVELS 4
#define POINTS 8
#define NQ    (BEV_H * BEV_W)          // 40000
#define NV    43520
#define NCLS  21
#define HW    40000

typedef __attribute__((ext_vector_type(16))) _Float16 v16h;
typedef __attribute__((ext_vector_type(8)))  float    v8f;
typedef __attribute__((ext_vector_type(4)))  unsigned int u32x4;
typedef __attribute__((ext_vector_type(8)))  int      i32x8;
typedef __attribute__((ext_vector_type(4)))  int      i32x4;

// =======================================================================
// WMMA GEMM with TDM-staged weights.
// Block = 256 threads (8 waves).  Block computes a 128(M) x 16(N) panel:
// wave w -> 16x16 tile at rows (blockIdx.y*8 + w)*16, cols blockIdx.x*16.
// The K x 16 f32 weight panel (K<=128 -> 8KB) is DMA'd to LDS once per
// block with tensor_load_to_lds (TDM, D# per ISA ch.8), then all waves
// consume it via DS reads.  A is read from global f32 and converted to
// f16 in registers; math is v_wmma_f32_16x16x32_f16.
// A frag layout (ISA 7.12.2): lane<16 -> row M=l, K={kc..kc+7,kc+16..kc+23};
// lane>=16 -> same row, K={kc+8..kc+15,kc+24..kc+31}.  B mirrors with N.
// C/D: VGPR r -> M = r + 8*(lane>=16), N = lane&15.
// =======================================================================
__global__ void gemm_wmma_tdm_kernel(const float* __restrict__ A,
                                     const float* __restrict__ W,
                                     const float* __restrict__ bias,
                                     float* __restrict__ C,
                                     int M, int N, int K, int relu) {
    __shared__ float shW[ED * 16];          // K x 16 weight panel (K <= 128)

    const int wave = threadIdx.x >> 5;
    const int lane = threadIdx.x & 31;
    const int half = lane >> 4;
    const int l    = lane & 15;
    const int c0   = blockIdx.x * 16;

    // ---- stage W[0:K, c0:c0+16] into LDS via the Tensor Data Mover ----
    if (wave == 0) {
        const unsigned long long ga =
            (unsigned long long)(const void*)(W + c0);   // tile start (bytes)
        // D# group 0: count=1, lds_addr=0, global_addr, type=2
        u32x4 g0;
        g0[0] = 1u;                                      // count = 1 valid D#
        g0[1] = 0u;                                      // lds_addr (bytes)
        g0[2] = (unsigned int)(ga & 0xFFFFFFFFull);
        g0[3] = (unsigned int)((ga >> 32) & 0x1FFFFFFull) | (2u << 30); // type=2
        // D# group 1: data_size=4B, tensor_dim0=N, tensor_dim1=K,
        //             tile_dim0=16, tile_dim1=K, dim0_stride=N
        i32x8 g1;
        g1[0] = (int)(2u << 16);                         // data_size code 2 = 4B
        g1[1] = (int)(((unsigned)N & 0xFFFFu) << 16);    // tensor_dim0 lo16
        g1[2] = (int)((((unsigned)N >> 16) & 0xFFFFu) |
                      (((unsigned)K & 0xFFFFu) << 16));  // dim0 hi16 | dim1 lo16
        g1[3] = (int)((((unsigned)K >> 16) & 0xFFFFu) |
                      (16u << 16));                      // dim1 hi16 | tile_dim0
        g1[4] = (int)((unsigned)K & 0xFFFFu);            // tile_dim1 (tile_dim2=0)
        g1[5] = (int)(unsigned)N;                        // dim0_stride lo32
        g1[6] = 0;                                       // stride hi16, dim1_stride=0
        g1[7] = 0;
        i32x4 gz  = {0, 0, 0, 0};                        // groups 2/3 unused (2D)
        i32x8 gz8 = {0, 0, 0, 0, 0, 0, 0, 0};            // trailing group (clang-23 form)
        __builtin_amdgcn_tensor_load_to_lds(g0, g1, gz, gz, gz8, 0);
        __builtin_amdgcn_s_wait_tensorcnt(0);
    }
    __syncthreads();

    const int mtiles = M >> 4;
    const int mt = blockIdx.y * 8 + wave;
    if (mt >= mtiles) return;                            // uniform per wave
    const int r0 = mt * 16;

    const float* arow = A + (size_t)(r0 + l) * K;
    const int n = c0 + l;

    v8f acc = {};
    for (int kc = 0; kc < K; kc += 32) {
        if (kc + 32 < K) __builtin_prefetch(arow + kc + 32, 0, 0);
        const int kb = kc + half * 8;
        v16h a, b;
#pragma unroll
        for (int j = 0; j < 8; ++j) {
            a[j]     = (_Float16)arow[kb + j];
            a[j + 8] = (_Float16)arow[kb + 16 + j];
            b[j]     = (_Float16)shW[(kb + j) * 16 + l];
            b[j + 8] = (_Float16)shW[(kb + 16 + j) * 16 + l];
        }
        acc = __builtin_amdgcn_wmma_f32_16x16x32_f16(
            false, a, false, b, (short)0, acc, false, false);
    }

    const float bv = bias ? bias[n] : 0.0f;
#pragma unroll
    for (int r = 0; r < 8; ++r) {
        float val = acc[r] + bv;
        if (relu) val = fmaxf(val, 0.0f);
        C[(size_t)(r0 + half * 8 + r) * N + n] = val;
    }
}

// =======================================================================
// WMMA convolution (NCHW, OIHW weights): M = 16 pixels, N = 16 out-chans,
// K = input channels (chunks of 32, zero-filled past Cin).  No bias
// (BatchNorm follows).  200x200 spatial, zero padding.
// =======================================================================
__global__ void conv_wmma_kernel(const float* __restrict__ in,
                                 const float* __restrict__ wgt,
                                 float* __restrict__ out,
                                 int Cin, int KH, int KW, int pad) {
    const int lane = threadIdx.x & 31;
    const int half = lane >> 4;
    const int l    = lane & 15;
    const int p0 = blockIdx.y * 16;     // pixel tile
    const int c0 = blockIdx.x * 16;     // out-channel tile

    const int pix = p0 + l;
    const int py = pix / BEV_W;
    const int px = pix % BEV_W;
    const int n  = c0 + l;              // out channel (B / store role)
    const int taps = KH * KW;

    v8f acc = {};
    for (int kh = 0; kh < KH; ++kh) {
        const int sy = py + kh - pad;
        const bool yok = (sy >= 0) && (sy < BEV_H);
        for (int kw = 0; kw < KW; ++kw) {
            const int sx = px + kw - pad;
            const bool ok = yok && (sx >= 0) && (sx < BEV_W);
            const float* ip = in + ((long)sy * BEV_W + sx);
            const float* wp = wgt + (size_t)n * Cin * taps + kh * KW + kw;
            for (int kc = 0; kc < Cin; kc += 32) {
                const int kb = kc + half * 8;
                v16h a, b;
#pragma unroll
                for (int j = 0; j < 8; ++j) {
                    const int ic0 = kb + j;
                    const int ic1 = kb + 16 + j;
                    a[j]     = (ok && ic0 < Cin) ? (_Float16)ip[(size_t)ic0 * HW] : (_Float16)0.0f;
                    a[j + 8] = (ok && ic1 < Cin) ? (_Float16)ip[(size_t)ic1 * HW] : (_Float16)0.0f;
                    b[j]     = (ic0 < Cin) ? (_Float16)wp[(size_t)ic0 * taps] : (_Float16)0.0f;
                    b[j + 8] = (ic1 < Cin) ? (_Float16)wp[(size_t)ic1 * taps] : (_Float16)0.0f;
                }
                acc = __builtin_amdgcn_wmma_f32_16x16x32_f16(
                    false, a, false, b, (short)0, acc, false, false);
            }
        }
    }
#pragma unroll
    for (int r = 0; r < 8; ++r) {
        const int pp = p0 + half * 8 + r;
        out[(size_t)n * HW + pp] = acc[r];
    }
}

// =======================================================================
// Deformable-attention sampling: one thread per (query, head).
// Softmax over 32 (level,point) logits, bilinear gather from v[NV,H,HD].
// =======================================================================
__global__ void deform_kernel(const float* __restrict__ v,
                              const float* __restrict__ offb,   // [NQ,512]
                              const float* __restrict__ attnl,  // [NQ,256]
                              float* __restrict__ out) {        // [NQ,ED]
    const int tid = blockIdx.x * blockDim.x + threadIdx.x;
    if (tid >= NQ * HEADS) return;
    const int q = tid >> 3;
    const int h = tid & 7;

    float logits[32];
    const float* al = attnl + (size_t)q * 256 + h * 32;
    float mx = -1e30f;
#pragma unroll
    for (int i = 0; i < 32; ++i) { logits[i] = al[i]; mx = fmaxf(mx, logits[i]); }
    float s = 0.0f;
#pragma unroll
    for (int i = 0; i < 32; ++i) { logits[i] = __expf(logits[i] - mx); s += logits[i]; }
    const float inv = 1.0f / s;

    const float refx = ((float)(q % BEV_W) + 0.5f) / (float)BEV_W;
    const float refy = ((float)(q / BEV_W) + 0.5f) / (float)BEV_H;

    float acc[HD];
#pragma unroll
    for (int d = 0; d < HD; ++d) acc[d] = 0.0f;

    const int lvlH[4] = {128, 64, 32, 16};
    const int lvlW[4] = {256, 128, 64, 32};
    const int lvlS[4] = {0, 32768, 40960, 43008};
    const float* offq = offb + (size_t)q * 512 + h * 64;

    for (int lvl = 0; lvl < LEVELS; ++lvl) {
        const int Wl = lvlW[lvl], Hl = lvlH[lvl], Sl = lvlS[lvl];
        for (int p = 0; p < POINTS; ++p) {
            const float aw = logits[lvl * POINTS + p] * inv;
            const float ox = offq[lvl * 16 + p * 2 + 0];
            const float oy = offq[lvl * 16 + p * 2 + 1];
            const float x = (refx + ox / (float)Wl) * (float)Wl - 0.5f;
            const float y = (refy + oy / (float)Hl) * (float)Hl - 0.5f;
            const float x0f = floorf(x), y0f = floorf(y);
            const int x0 = (int)x0f, y0 = (int)y0f;
            const float fx = x - x0f, fy = y - y0f;
#pragma unroll
            for (int c = 0; c < 4; ++c) {
                const int dx = c & 1, dy = c >> 1;
                const int xc = x0 + dx, yc = y0 + dy;
                const float w = (dx ? fx : 1.0f - fx) * (dy ? fy : 1.0f - fy);
                if (xc < 0 || xc >= Wl || yc < 0 || yc >= Hl) continue;
                const float* vp = v + (size_t)(Sl + yc * Wl + xc) * ED + h * HD;
                const float wa = aw * w;
#pragma unroll
                for (int d = 0; d < HD; ++d) acc[d] += wa * vp[d];
            }
        }
    }
    float* op = out + (size_t)q * ED + h * HD;
#pragma unroll
    for (int d = 0; d < HD; ++d) op[d] = acc[d];
}

// ---------------- LayerNorm(residual) : block(128) per row -------------
__global__ void ln_res_kernel(const float* __restrict__ x,
                              const float* __restrict__ resid,
                              const float* __restrict__ g,
                              const float* __restrict__ b,
                              float* __restrict__ out) {
    const int row = blockIdx.x;
    const int t = threadIdx.x;
    const float v = x[(size_t)row * ED + t] + resid[(size_t)row * ED + t];
    __shared__ float sh[ED];
    sh[t] = v; __syncthreads();
    for (int st = ED / 2; st > 0; st >>= 1) {
        if (t < st) sh[t] += sh[t + st];
        __syncthreads();
    }
    const float mean = sh[0] / (float)ED;
    __syncthreads();
    const float d = v - mean;
    sh[t] = d * d; __syncthreads();
    for (int st = ED / 2; st > 0; st >>= 1) {
        if (t < st) sh[t] += sh[t + st];
        __syncthreads();
    }
    const float var = sh[0] / (float)ED;
    out[(size_t)row * ED + t] = d * rsqrtf(var + 1e-5f) * g[t] + b[t];
}

// ---------------- misc elementwise / reduction kernels -----------------
__global__ void add_kernel(const float* a, const float* b, float* o, int n) {
    int i = blockIdx.x * blockDim.x + threadIdx.x;
    if (i < n) o[i] = a[i] + b[i];
}

__global__ void transpose_kernel(const float* __restrict__ q, float* __restrict__ mem) {
    int i = blockIdx.x * blockDim.x + threadIdx.x;     // i = c*HW + p
    if (i >= ED * HW) return;
    int c = i / HW, p = i % HW;
    mem[i] = q[(size_t)p * ED + c];
}

__global__ void bn_stats_kernel(const float* __restrict__ in, float* __restrict__ stats) {
    const int c = blockIdx.x;
    const float* p = in + (size_t)c * HW;
    float s = 0.0f, s2 = 0.0f;
    for (int i = threadIdx.x; i < HW; i += blockDim.x) {
        float v = p[i]; s += v; s2 += v * v;
    }
    __shared__ float sh[256], sh2[256];
    sh[threadIdx.x] = s; sh2[threadIdx.x] = s2; __syncthreads();
    for (int st = blockDim.x / 2; st > 0; st >>= 1) {
        if ((int)threadIdx.x < st) { sh[threadIdx.x] += sh[threadIdx.x + st]; sh2[threadIdx.x] += sh2[threadIdx.x + st]; }
        __syncthreads();
    }
    if (threadIdx.x == 0) {
        float m = sh[0] / (float)HW;
        stats[c * 2] = m;
        stats[c * 2 + 1] = sh2[0] / (float)HW - m * m;
    }
}

__global__ void bn_relu_kernel(const float* __restrict__ in,
                               const float* __restrict__ stats,
                               const float* __restrict__ g,
                               const float* __restrict__ b,
                               float* __restrict__ out, int C) {
    int i = blockIdx.x * blockDim.x + threadIdx.x;
    if (i >= C * HW) return;
    int c = i / HW;
    float m = stats[c * 2], var = stats[c * 2 + 1];
    float v = (in[i] - m) * rsqrtf(var + 1e-5f) * g[c] + b[c];
    out[i] = fmaxf(v, 0.0f);
}

__global__ void objconv_kernel(const float* __restrict__ h,
                               const float* __restrict__ w,
                               const float* __restrict__ bias,
                               float* __restrict__ semmap) {
    int i = blockIdx.x * blockDim.x + threadIdx.x;     // i = oc*HW + p
    if (i >= NCLS * HW) return;
    int oc = i / HW, p = i % HW;
    float acc = bias[oc];
#pragma unroll
    for (int ic = 0; ic < 48; ++ic) acc += h[(size_t)ic * HW + p] * w[oc * 48 + ic];
    semmap[i] = acc;
}

__global__ void max_init_kernel(int* slot) { if (threadIdx.x == 0) *slot = INT_MIN; }

__global__ void max_reduce_kernel(const int* __restrict__ proj, int* slot, int n) {
    int i = blockIdx.x * blockDim.x + threadIdx.x;
    int v = (i < n) ? proj[i] : INT_MIN;
    __shared__ int sh[256];
    sh[threadIdx.x] = v; __syncthreads();
    for (int st = blockDim.x / 2; st > 0; st >>= 1) {
        if ((int)threadIdx.x < st) sh[threadIdx.x] = max(sh[threadIdx.x], sh[threadIdx.x + st]);
        __syncthreads();
    }
    if (threadIdx.x == 0) atomicMax(slot, sh[0]);
}

__global__ void mask_kernel(const int* __restrict__ proj, const int* slot,
                            float* __restrict__ out, int n) {
    int i = blockIdx.x * blockDim.x + threadIdx.x;
    if (i < n) out[i] = (proj[i] < *slot) ? 1.0f : 0.0f;
}

// =======================================================================
// host-side launch
// =======================================================================
extern "C" void kernel_launch(void* const* d_in, const int* in_sizes, int n_in,
                              void* d_out, int out_size, void* d_ws, size_t ws_size,
                              hipStream_t stream) {
    // -------- inputs (setup_inputs order) --------
    const float* value  = (const float*)d_in[0];
    const float* bev_q  = (const float*)d_in[1];
    const float* bev_p  = (const float*)d_in[2];
    const int*   proj   = (const int*)d_in[3];
    const float* Wv     = (const float*)d_in[4];
    const float* bv     = (const float*)d_in[5];
    const float* Woff   = (const float*)d_in[6];
    const float* boff   = (const float*)d_in[7];
    const float* Wattn  = (const float*)d_in[8];
    const float* battn  = (const float*)d_in[9];
    const float* Wout   = (const float*)d_in[10];
    const float* bout   = (const float*)d_in[11];
    const float* Wf1    = (const float*)d_in[12];
    const float* bf1    = (const float*)d_in[13];
    const float* Wf2    = (const float*)d_in[14];
    const float* bf2    = (const float*)d_in[15];
    const float* ln_g   = (const float*)d_in[16];
    const float* ln_b   = (const float*)d_in[17];
    const float* cw1 = (const float*)d_in[18]; const float* g1 = (const float*)d_in[19]; const float* b1 = (const float*)d_in[20];
    const float* cw2 = (const float*)d_in[21]; const float* g2 = (const float*)d_in[22]; const float* b2 = (const float*)d_in[23];
    const float* cw3 = (const float*)d_in[24]; const float* g3 = (const float*)d_in[25]; const float* b3 = (const float*)d_in[26];
    const float* cw4 = (const float*)d_in[27]; const float* g4 = (const float*)d_in[28]; const float* b4 = (const float*)d_in[29];
    const float* objw = (const float*)d_in[30]; const float* objb = (const float*)d_in[31];

    float* out = (float*)d_out;                // [0,840000) semmap, [840000,880000) mask

    // -------- workspace layout (floats) --------
    float* wsf = (float*)d_ws;
    const size_t O_Q    = 0;                          // NQ*ED
    const size_t O_QPOS = O_Q    + (size_t)NQ * ED;   // NQ*ED
    const size_t O_V    = O_QPOS + (size_t)NQ * ED;   // NV*ED
    const size_t O_OFF  = O_V    + (size_t)NV * ED;   // NQ*512
    const size_t O_ATT  = O_OFF  + (size_t)NQ * 512;  // NQ*256
    const size_t O_T0   = O_ATT  + (size_t)NQ * 256;  // NQ*ED
    const size_t O_T1   = O_T0   + (size_t)NQ * ED;   // NQ*ED
    const size_t O_ST   = O_T1   + (size_t)NQ * ED;   // 2*128 stats
    const size_t O_MAX  = O_ST   + 256;               // 1 int slot
    // conv-phase reuse of encoder scratch:
    float* memb = wsf + O_QPOS;   // [128,200,200]
    float* c1   = wsf + O_V;      // [128,200,200]
    float* c2   = wsf + O_OFF;    // [64,200,200]
    float* c3   = wsf + O_ATT;    // [48,200,200]
    float* c4   = wsf + O_T0;     // [48,200,200]
    float* stats = wsf + O_ST;
    int*   maxslot = (int*)(wsf + O_MAX);

    const dim3 B32(32), B128(128), B256(256);
    const int MT_NQ = (NQ / 16 + 7) / 8;   // 313 row-groups of 8 tiles
    const int MT_NV = (NV / 16 + 7) / 8;   // 340

    // -------- observed_masks --------
    max_init_kernel<<<1, 32, 0, stream>>>(maxslot);
    max_reduce_kernel<<<(HW + 255) / 256, B256, 0, stream>>>(proj, maxslot, HW);
    mask_kernel<<<(HW + 255) / 256, B256, 0, stream>>>(proj, maxslot, out + (size_t)NCLS * HW, HW);

    // -------- encoder --------
    (void)hipMemcpyAsync(wsf + O_Q, bev_q, (size_t)NQ * ED * sizeof(float),
                         hipMemcpyDeviceToDevice, stream);

    for (int i = 0; i < 2; ++i) {
        const float* Wv_i   = Wv   + (size_t)i * ED * ED;
        const float* Woff_i = Woff + (size_t)i * ED * 512;
        const float* Wat_i  = Wattn+ (size_t)i * ED * 256;
        const float* Wo_i   = Wout + (size_t)i * ED * ED;
        const float* W1_i   = Wf1  + (size_t)i * ED * ED;
        const float* W2_i   = Wf2  + (size_t)i * ED * ED;

        // qpos = q + bev_pos
        add_kernel<<<(NQ * ED + 255) / 256, B256, 0, stream>>>(wsf + O_Q, bev_p, wsf + O_QPOS, NQ * ED);
        // v = value @ Wv + bv          [NV,128]
        gemm_wmma_tdm_kernel<<<dim3(ED / 16, MT_NV), B256, 0, stream>>>(
            value, Wv_i, bv + i * ED, wsf + O_V, NV, ED, ED, 0);
        // off = qpos @ Woff + boff     [NQ,512]
        gemm_wmma_tdm_kernel<<<dim3(512 / 16, MT_NQ), B256, 0, stream>>>(
            wsf + O_QPOS, Woff_i, boff + i * 512, wsf + O_OFF, NQ, 512, ED, 0);
        // attn logits = qpos @ Wattn   [NQ,256]
        gemm_wmma_tdm_kernel<<<dim3(256 / 16, MT_NQ), B256, 0, stream>>>(
            wsf + O_QPOS, Wat_i, battn + i * 256, wsf + O_ATT, NQ, 256, ED, 0);
        // sampling -> t0 [NQ,128]
        deform_kernel<<<(NQ * HEADS + 255) / 256, B256, 0, stream>>>(
            wsf + O_V, wsf + O_OFF, wsf + O_ATT, wsf + O_T0);
        // a = t0 @ Wout + bout -> t1
        gemm_wmma_tdm_kernel<<<dim3(ED / 16, MT_NQ), B256, 0, stream>>>(
            wsf + O_T0, Wo_i, bout + i * ED, wsf + O_T1, NQ, ED, ED, 0);
        // q = LN(q + a)
        ln_res_kernel<<<NQ, B128, 0, stream>>>(
            wsf + O_Q, wsf + O_T1, ln_g + (size_t)(i * 2 + 0) * ED, ln_b + (size_t)(i * 2 + 0) * ED, wsf + O_Q);
        // h1 = relu(q @ Wf1 + bf1) -> t0
        gemm_wmma_tdm_kernel<<<dim3(ED / 16, MT_NQ), B256, 0, stream>>>(
            wsf + O_Q, W1_i, bf1 + i * ED, wsf + O_T0, NQ, ED, ED, 1);
        // h2 = h1 @ Wf2 + bf2 -> t1
        gemm_wmma_tdm_kernel<<<dim3(ED / 16, MT_NQ), B256, 0, stream>>>(
            wsf + O_T0, W2_i, bf2 + i * ED, wsf + O_T1, NQ, ED, ED, 0);
        // q = LN(q + h2)
        ln_res_kernel<<<NQ, B128, 0, stream>>>(
            wsf + O_Q, wsf + O_T1, ln_g + (size_t)(i * 2 + 1) * ED, ln_b + (size_t)(i * 2 + 1) * ED, wsf + O_Q);
    }

    // -------- conv head --------
    transpose_kernel<<<(ED * HW + 255) / 256, B256, 0, stream>>>(wsf + O_Q, memb);

    conv_wmma_kernel<<<dim3(128 / 16, HW / 16), B32, 0, stream>>>(memb, cw1, c1, 128, 7, 7, 3);
    bn_stats_kernel<<<128, B256, 0, stream>>>(c1, stats);
    bn_relu_kernel<<<(128 * HW + 255) / 256, B256, 0, stream>>>(c1, stats, g1, b1, c1, 128);

    conv_wmma_kernel<<<dim3(64 / 16, HW / 16), B32, 0, stream>>>(c1, cw2, c2, 128, 3, 3, 1);
    bn_stats_kernel<<<64, B256, 0, stream>>>(c2, stats);
    bn_relu_kernel<<<(64 * HW + 255) / 256, B256, 0, stream>>>(c2, stats, g2, b2, c2, 64);

    conv_wmma_kernel<<<dim3(48 / 16, HW / 16), B32, 0, stream>>>(c2, cw3, c3, 64, 3, 3, 1);
    bn_stats_kernel<<<48, B256, 0, stream>>>(c3, stats);
    bn_relu_kernel<<<(48 * HW + 255) / 256, B256, 0, stream>>>(c3, stats, g3, b3, c3, 48);

    conv_wmma_kernel<<<dim3(48 / 16, HW / 16), B32, 0, stream>>>(c3, cw4, c4, 48, 3, 3, 1);
    bn_stats_kernel<<<48, B256, 0, stream>>>(c4, stats);
    bn_relu_kernel<<<(48 * HW + 255) / 256, B256, 0, stream>>>(c4, stats, g4, b4, c4, 48);

    objconv_kernel<<<(NCLS * HW + 255) / 256, B256, 0, stream>>>(c4, objw, objb, out);
}
